// LinearAttention_61692910240501
// MI455X (gfx1250) — matrix operand order repaired and checked by reading, not compiled, not run on hardware
//
#include <hip/hip_runtime.h>

typedef __bf16 v16bf __attribute__((ext_vector_type(16)));
typedef __bf16 v8bf  __attribute__((ext_vector_type(8)));
typedef __bf16 v4bf  __attribute__((ext_vector_type(4)));
typedef float  v8f   __attribute__((ext_vector_type(8)));

union Frag16 { v8bf h[2]; v16bf v; };  // two b128 halves <-> one 16xbf16 fragment

#define NBH 32      // B*H
#define DDIM 64

__device__ __forceinline__ float phi_elu1(float x) {
    // elu(x)+1 : x>0 -> x+1 ; x<=0 -> exp(x)
    return x > 0.0f ? x + 1.0f : __expf(x);
}

// ---------------------------------------------------------------- k0: zero ws
__global__ void __launch_bounds__(256) la_zero_ws(float* __restrict__ ws, int n) {
    int i = blockIdx.x * 256 + threadIdx.x;
    if (i < n) ws[i] = 0.0f;
}

// ---------------------------------------------------------------- k1: kv & ksum
// grid.x = NBH * (L/256); block = 128 (4 waves)
// kv[d,e] += sum_l phiK[l,d]*V[l,e]  via v_wmma_f32_16x16x32_bf16
// LDS tiles stored TRANSPOSED ([d][l], [e][l]) so fragment halves are
// 8 contiguous bf16 -> single ds_load_b128 each.
__global__ void __launch_bounds__(128) la_kv_kernel(
        const float* __restrict__ Kp, const float* __restrict__ Vp,
        float* __restrict__ kv_ws, float* __restrict__ ksum_ws) {
    __shared__ __align__(16) __bf16 sKt[64][40];   // [d][l]
    __shared__ __align__(16) __bf16 sVt[64][40];   // [e][l]

    const int bh    = blockIdx.x & (NBH - 1);
    const int chunk = blockIdx.x >> 5;
    const int tid   = threadIdx.x;
    const int lane  = tid & 31;
    const int w     = tid >> 5;          // wave 0..3 -> rows 16w..16w+15
    const int hl    = lane >> 4;         // wave half
    const int lm    = lane & 15;
    const int dg    = 4 * (tid & 15);    // this thread's fixed 4-d group
    const int lb8   = tid >> 4;          // 0..7

    float ksum_acc[4] = {0.f, 0.f, 0.f, 0.f};
    v8f acc[4];
#pragma unroll
    for (int t = 0; t < 4; ++t)
#pragma unroll
        for (int r = 0; r < 8; ++r) acc[t][r] = 0.0f;

    for (int it = 0; it < 8; ++it) {             // 8 * 32 = 256 l's per block
        const int l0 = chunk * 256 + it * 32;
        // ---- stage 32x64 tiles with float4 (b128) loads, transpose into LDS
#pragma unroll
        for (int p = 0; p < 4; ++p) {
            int l = lb8 + 8 * p;
            size_t g = ((size_t)(l0 + l) * NBH + bh) * DDIM + dg;
            float4 kf = *(const float4*)(Kp + g);
            float4 vf = *(const float4*)(Vp + g);
            float pk[4] = { phi_elu1(kf.x), phi_elu1(kf.y),
                            phi_elu1(kf.z), phi_elu1(kf.w) };
            float vv[4] = { vf.x, vf.y, vf.z, vf.w };
#pragma unroll
            for (int j = 0; j < 4; ++j) {
                ksum_acc[j] += pk[j];
                sKt[dg + j][l] = (__bf16)pk[j];   // native v_cvt to bf16
                sVt[dg + j][l] = (__bf16)vv[j];
            }
        }
        __syncthreads();

        // ---- A fragment: A[m,k] = phiK[l=k, d=16w+m]; halves contiguous
        Frag16 a;
        const int row = 16 * w + lm;
        a.h[0] = *(const v8bf*)&sKt[row][8 * hl];
        a.h[1] = *(const v8bf*)&sKt[row][16 + 8 * hl];
        // ---- B fragments: B[k,n] = V[l=k, e=16t+n]
#pragma unroll
        for (int t = 0; t < 4; ++t) {
            Frag16 b;
            b.h[0] = *(const v8bf*)&sVt[16 * t + lm][8 * hl];
            b.h[1] = *(const v8bf*)&sVt[16 * t + lm][16 + 8 * hl];
            acc[t] = __builtin_amdgcn_wmma_f32_16x16x32_bf16(
                false, a.v, false, b.v, (short)0, acc[t], false, false);
        }
        __syncthreads();
    }

    // ---- accumulate partial 64x64 kv tile into global ws
#pragma unroll
    for (int t = 0; t < 4; ++t)
#pragma unroll
        for (int r = 0; r < 8; ++r) {
            int m = 16 * w + r + 8 * hl;
            int n = 16 * t + lm;
            atomicAdd(&kv_ws[((size_t)bh * DDIM + m) * DDIM + n], acc[t][r]);
        }
#pragma unroll
    for (int j = 0; j < 4; ++j)
        atomicAdd(&ksum_ws[bh * DDIM + dg + j], ksum_acc[j]);
}

// ---------------------------------------------------------------- k2: output
// grid.x = NBH * (L/128); block = 256 (8 waves, 16 l-rows each)
// Denominator via an extra WMMA with ksum broadcast into all 16 B-columns:
// D[m,n] = den[m] for every n -> accumulator VGPR r holds den for its row.
__global__ void __launch_bounds__(256) la_out_kernel(
        const float* __restrict__ Qp,
        const float* __restrict__ kv_ws, const float* __restrict__ ksum_ws,
        float* __restrict__ out) {
    __shared__ __align__(16) __bf16 skvT[64][72];  // kv [e][d]
    __shared__ __align__(16) __bf16 sq[128][72];   // phiQ [row][d]
    __shared__ __align__(16) __bf16 sksb[64];      // ksum

    const int bh   = blockIdx.x & (NBH - 1);
    const int lb   = blockIdx.x >> 5;
    const int tid  = threadIdx.x;
    const int lane = tid & 31;
    const int w    = tid >> 5;
    const int hl   = lane >> 4;
    const int lm   = lane & 15;
    const int eg   = 4 * (tid & 15);
    const int t4   = tid >> 4;           // 0..15

    // stage kv (L2-resident ws) transposed [e][d]
#pragma unroll
    for (int p = 0; p < 4; ++p) {
        int d = t4 + 16 * p;
        float4 kvf = *(const float4*)(kv_ws + (size_t)bh * 4096 + d * 64 + eg);
        skvT[eg + 0][d] = (__bf16)kvf.x;
        skvT[eg + 1][d] = (__bf16)kvf.y;
        skvT[eg + 2][d] = (__bf16)kvf.z;
        skvT[eg + 3][d] = (__bf16)kvf.w;
    }
    // stage 128x64 phiQ tile, [row][d] (fragment-major for A), b64 stores
#pragma unroll
    for (int p = 0; p < 8; ++p) {
        int row = t4 + 16 * p;
        size_t g = ((size_t)(lb * 128 + row) * NBH + bh) * DDIM + eg;
        float4 qf = *(const float4*)(Qp + g);
        v4bf s;
        s[0] = (__bf16)phi_elu1(qf.x);
        s[1] = (__bf16)phi_elu1(qf.y);
        s[2] = (__bf16)phi_elu1(qf.z);
        s[3] = (__bf16)phi_elu1(qf.w);
        *(v4bf*)&sq[row][eg] = s;
    }
    if (tid < 64) sksb[tid] = (__bf16)ksum_ws[bh * DDIM + tid];
    __syncthreads();

    // A fragments for this wave's 16 rows; K = d in [0,32) / [32,64)
    const int rowt = 16 * w + lm;
    Frag16 a0, a1;
    a0.h[0] = *(const v8bf*)&sq[rowt][8 * hl];
    a0.h[1] = *(const v8bf*)&sq[rowt][16 + 8 * hl];
    a1.h[0] = *(const v8bf*)&sq[rowt][32 + 8 * hl];
    a1.h[1] = *(const v8bf*)&sq[rowt][48 + 8 * hl];

    // ksum broadcast B fragments (lane-uniform per half-wave)
    Frag16 db0, db1;
    db0.h[0] = *(const v8bf*)&sksb[8 * hl];
    db0.h[1] = *(const v8bf*)&sksb[16 + 8 * hl];
    db1.h[0] = *(const v8bf*)&sksb[32 + 8 * hl];
    db1.h[1] = *(const v8bf*)&sksb[48 + 8 * hl];

    v8f acc[4], dacc;
#pragma unroll
    for (int r = 0; r < 8; ++r) dacc[r] = 0.0f;
#pragma unroll
    for (int t = 0; t < 4; ++t)
#pragma unroll
        for (int r = 0; r < 8; ++r) acc[t][r] = 0.0f;

    dacc = __builtin_amdgcn_wmma_f32_16x16x32_bf16(
        false, a0.v, false, db0.v, (short)0, dacc, false, false);
    dacc = __builtin_amdgcn_wmma_f32_16x16x32_bf16(
        false, a1.v, false, db1.v, (short)0, dacc, false, false);

#pragma unroll
    for (int t = 0; t < 4; ++t) {
        Frag16 b0, b1;
        b0.h[0] = *(const v8bf*)&skvT[16 * t + lm][8 * hl];
        b0.h[1] = *(const v8bf*)&skvT[16 * t + lm][16 + 8 * hl];
        b1.h[0] = *(const v8bf*)&skvT[16 * t + lm][32 + 8 * hl];
        b1.h[1] = *(const v8bf*)&skvT[16 * t + lm][48 + 8 * hl];
        acc[t] = __builtin_amdgcn_wmma_f32_16x16x32_bf16(
            false, a0.v, false, b0.v, (short)0, acc[t], false, false);
        acc[t] = __builtin_amdgcn_wmma_f32_16x16x32_bf16(
            false, a1.v, false, b1.v, (short)0, acc[t], false, false);
    }

    // divide by denominator and store: VGPR r -> row = r + 8*hl, col = lm
#pragma unroll
    for (int r = 0; r < 8; ++r) {
        float invd = 1.0f / (dacc[r] + 1e-6f);
        int l = lb * 128 + 16 * w + r + 8 * hl;
#pragma unroll
        for (int t = 0; t < 4; ++t) {
            size_t g = ((size_t)l * NBH + bh) * DDIM + 16 * t + lm;
            out[g] = acc[t][r] * invd;
        }
    }
}

// ---------------------------------------------------------------- launch
extern "C" void kernel_launch(void* const* d_in, const int* in_sizes, int n_in,
                              void* d_out, int out_size, void* d_ws, size_t ws_size,
                              hipStream_t stream) {
    const float* Q = (const float*)d_in[0];
    const float* K = (const float*)d_in[1];
    const float* V = (const float*)d_in[2];
    float* out = (float*)d_out;

    float* kv_ws   = (float*)d_ws;               // 32*64*64 f32
    float* ksum_ws = kv_ws + NBH * DDIM * DDIM;  // 32*64 f32
    const int nzero = NBH * DDIM * DDIM + NBH * DDIM;   // 133120

    const int L = in_sizes[0] / (NBH * DDIM);    // 4096

    la_zero_ws<<<(nzero + 255) / 256, 256, 0, stream>>>(kv_ws, nzero);
    la_kv_kernel<<<NBH * (L / 256), 128, 0, stream>>>(K, V, kv_ws, ksum_ws);
    la_out_kernel<<<NBH * (L / 128), 256, 0, stream>>>(Q, kv_ws, ksum_ws, out);
}
